// NonlocalBlock_893353198276
// MI455X (gfx1250) — matrix operand order repaired and checked by reading, not compiled, not run on hardware
//
#include <hip/hip_runtime.h>
#include <stdint.h>

// Problem constants (from reference): B=4, C=256, W=H=64 -> N=4096, CQ=32.
#define B_   4
#define C_   256
#define N_   4096
#define CQ_  32
#define DTOT 320   // stacked output channels: 32 (q) + 32 (k) + 256 (v)
#define XPAD 264   // xs row stride (elements): 256 + 8 pad (16B multiple, bank-skewed)

typedef __attribute__((ext_vector_type(16))) __bf16         v16bf;
typedef __attribute__((ext_vector_type(8)))  float          v8f;
typedef __attribute__((ext_vector_type(16))) unsigned short v16u;
typedef __attribute__((ext_vector_type(8)))  unsigned short su8;

// Native f32 -> bf16 (lowers to v_cvt_pk_bf16_f32 on gfx1250).
static __device__ inline unsigned short f2bfu(float f) {
  return __builtin_bit_cast(unsigned short, (__bf16)f);
}

// Two f32 -> packed 2x bf16 in one dword (one v_cvt_pk_bf16_f32).
static __device__ inline unsigned pk2(float a, float b) {
  return (unsigned)f2bfu(a) | ((unsigned)f2bfu(b) << 16);
}

static __device__ inline v16bf pack16(su8 lo, su8 hi) {
  v16u t;
#pragma unroll
  for (int i = 0; i < 8; ++i) { t[i] = lo[i]; t[8 + i] = hi[i]; }
  return __builtin_bit_cast(v16bf, t);
}

static __device__ inline v8f wmma_bf16(v16bf a, v16bf b, v8f c) {
  return __builtin_amdgcn_wmma_f32_16x16x32_bf16(false, a, false, b,
                                                 (short)0, c, false, false);
}

// gfx1250 async global->LDS copy (ASYNCcnt-tracked), per CDNA5 ISA
static __device__ inline void async_b128(unsigned lds_addr, const void* gaddr) {
  asm volatile("global_load_async_to_lds_b128 %0, %1, off"
               :: "v"(lds_addr), "v"(gaddr) : "memory");
}

// ---------------------------------------------------------------------------
// Kernel 0: one-shot weight conversion to bf16, stacked [Wq;Wk;Wv] = [320][256].
// ---------------------------------------------------------------------------
__global__ __launch_bounds__(256) void wcvt_kernel(
    const float* __restrict__ Wq, const float* __restrict__ Wk,
    const float* __restrict__ Wv, unsigned short* __restrict__ Wc)
{
  const int i = (blockIdx.x * 256 + threadIdx.x) * 4;   // 81920 elements total
  if (i >= DTOT * C_) return;
  const int row = i / C_;
  const int col = i % C_;
  const float* src;
  if (row < 32)      src = Wq + (size_t)row * C_;
  else if (row < 64) src = Wk + (size_t)(row - 32) * C_;
  else               src = Wv + (size_t)(row - 64) * C_;
  const float4 v = *(const float4*)(src + col);
  unsigned* dst = (unsigned*)(Wc + i);
  dst[0] = pk2(v.x, v.y);
  dst[1] = pk2(v.z, v.w);
}

// ---------------------------------------------------------------------------
// Kernel 1: fused QKV projection.  A workgroup owns (batch, 64-wide n chunk),
// stages x transposed to LDS as bf16 [n][c], then computes
// Y[320 x 64] = Wc[320 x 256] * x[256 x 64] + bias with bf16 WMMAs.
// Store region (Q/K/V) is uniform per 16-row d-tile -> scalar branch only.
// Outputs (bf16): Q [B][N][32], K(^T) [B][N][32], V [B][C][N].
// ---------------------------------------------------------------------------
__global__ __launch_bounds__(256) void qkv_proj_kernel(
    const float* __restrict__ x,
    const float* __restrict__ bq, const float* __restrict__ bk,
    const float* __restrict__ bv,
    const unsigned short* __restrict__ Wc,
    unsigned short* __restrict__ Qb, unsigned short* __restrict__ Kb,
    unsigned short* __restrict__ Vb)
{
  __shared__ __attribute__((aligned(16))) unsigned short xs[64][XPAD]; // ~33 KB

  const int tid = threadIdx.x;
  const int b   = blockIdx.x >> 6;           // 4 batches
  const int n0  = (blockIdx.x & 63) * 64;    // 64-wide spatial chunk

  // Stage x[b][:, n0..n0+63] -> xs[n][c] (bf16, transposed, 2 channels/store).
  const float* xb = x + (size_t)b * C_ * N_;
  {
    const int nl = tid & 63;                 // n within chunk
    const int cb = (tid >> 6) * 2;           // 0,2,4,6
#pragma unroll 1
    for (int c = cb; c < C_; c += 8) {
      const float v0 = xb[(size_t)c * N_ + n0 + nl];
      const float v1 = xb[(size_t)(c + 1) * N_ + n0 + nl];
      *(unsigned*)&xs[nl][c] = pk2(v0, v1);
    }
  }
  __syncthreads();

  const int lane  = tid & 31;
  const int w     = tid >> 5;
  const int lr    = lane & 15;
  const int hi16  = (lane >= 16);
  const int half8 = hi16 ? 8 : 0;

  // Wave w owns d-tiles {w, w+8, w+16 (<20)}; 4 n-subtiles each.
#pragma unroll 1
  for (int dt = w; dt < DTOT / 16; dt += 8) {
    const int d0 = dt * 16;
    const float* bias; int dbase;
    if (d0 < 32)      { bias = bq; dbase = 0;  }
    else if (d0 < 64) { bias = bk; dbase = 32; }
    else              { bias = bv; dbase = 64; }

    v8f acc[4];
#pragma unroll
    for (int nt = 0; nt < 4; ++nt)
#pragma unroll
      for (int j = 0; j < 8; ++j) acc[nt][j] = bias[d0 - dbase + j + half8];

    const su8* ap0 = (const su8*)(Wc + (size_t)(d0 + lr) * C_);
#pragma unroll 1
    for (int kc = 0; kc < C_; kc += 32) {
      const su8* ap = ap0 + (kc >> 3);
      const v16bf A = pack16(ap[hi16 ? 2 : 0], ap[hi16 ? 3 : 1]);
#pragma unroll
      for (int nt = 0; nt < 4; ++nt) {
        const su8* bp = (const su8*)&xs[nt * 16 + lr][kc + (hi16 ? 16 : 0)];
        const v16bf Bv = pack16(bp[0], bp[1]);
        acc[nt] = wmma_bf16(A, Bv, acc[nt]);
      }
    }

    // Store. Region is uniform per tile: single scalar branch, no exec churn.
    if (d0 < 64) {
      // Q or K region: [B][N][32] layout, (j, j+1) adjacent -> packed 4B stores.
      unsigned short* dst = (d0 < 32) ? Qb : Kb;
      const int dq = (d0 & 31) + half8;
#pragma unroll
      for (int nt = 0; nt < 4; ++nt) {
        const int n = n0 + nt * 16 + lr;
        unsigned short* row = dst + ((size_t)b * N_ + n) * CQ_ + dq;
#pragma unroll
        for (int j = 0; j < 8; j += 2)
          *(unsigned*)(row + j) = pk2(acc[nt][j], acc[nt][j + 1]);
      }
    } else {
      // V region: [B][C][N] layout, lanes contiguous along n -> coalesced b16.
      const int c0 = d0 - 64 + half8;
#pragma unroll
      for (int nt = 0; nt < 4; ++nt) {
        const int n = n0 + nt * 16 + lr;
        unsigned short* col = Vb + ((size_t)b * C_ + c0) * N_ + n;
#pragma unroll
        for (int j = 0; j < 8; ++j)
          col[(size_t)j * N_] = f2bfu(acc[nt][j]);
      }
    }
  }
}

// ---------------------------------------------------------------------------
// Kernel 2: fused flash attention + residual.
// 8 waves/WG; wave w owns query tile [n0..n0+15], accumulates O^T = V * P^T
// over 128 key blocks of 32 columns. V blocks staged via async->LDS double
// buffer shared by the workgroup; 18 WMMAs per key block.
// ---------------------------------------------------------------------------
__global__ __launch_bounds__(256) void attn_kernel(
    const float* __restrict__ x, const float* __restrict__ gamma,
    const unsigned short* __restrict__ Qb, const unsigned short* __restrict__ Kb,
    const unsigned short* __restrict__ Vb, float* __restrict__ out)
{
  __shared__ __attribute__((aligned(16))) unsigned short Vbuf[2][C_][32]; // 32 KB
  __shared__ __attribute__((aligned(16))) unsigned short Pbuf[8][16][32]; //  8 KB
  __shared__ float rowbuf[8][16];

  const int lane = threadIdx.x & 31;
  const int w    = threadIdx.x >> 5;
  const int lr   = lane & 15;
  const int hi16 = (lane >= 16);

  const int qt = blockIdx.x * 8 + w;      // 1024 query tiles total
  const int b  = qt >> 8;
  const int n0 = (qt & 255) * 16;

  // Q A-operand (held in registers for the whole kernel).
  const su8* qp = (const su8*)(Qb + ((size_t)b * N_ + n0 + lr) * CQ_);
  const v16bf qa = pack16(qp[hi16 ? 2 : 0], qp[hi16 ? 3 : 1]);

  v8f O[16];
#pragma unroll
  for (int t = 0; t < 16; ++t)
#pragma unroll
    for (int j = 0; j < 8; ++j) O[t][j] = 0.0f;

  float mx[8], lp[8];
#pragma unroll
  for (int j = 0; j < 8; ++j) { mx[j] = -3.0e38f; lp[j] = 0.0f; }

  const unsigned short* Vbb = Vb + (size_t)b * C_ * N_;
  const unsigned short* Kbb = Kb + (size_t)b * N_ * CQ_;

  // Prefetch V block 0 into LDS buffer 0 (each thread owns one channel row).
  {
    const unsigned lds = (unsigned)(uintptr_t)&Vbuf[0][threadIdx.x][0];
    const char* g = (const char*)(Vbb + (size_t)threadIdx.x * N_);
#pragma unroll
    for (int i = 0; i < 4; ++i) async_b128(lds + 16 * i, g + 16 * i);
  }

  const int NB = N_ / 32;                  // 128 key blocks
#pragma unroll 1
  for (int blk = 0; blk < NB; ++blk) {
    const int cur = blk & 1, nxt = cur ^ 1;
    const int m0  = blk * 32;

    if (blk + 1 < NB) {
      const unsigned lds = (unsigned)(uintptr_t)&Vbuf[nxt][threadIdx.x][0];
      const char* g = (const char*)(Vbb + (size_t)threadIdx.x * N_ + (m0 + 32));
#pragma unroll
      for (int i = 0; i < 4; ++i) async_b128(lds + 16 * i, g + 16 * i);
      __builtin_prefetch(Kbb + (size_t)(m0 + 32) * CQ_, 0, 0);
      asm volatile("s_wait_asynccnt 4" ::: "memory");  // current block landed
    } else {
      asm volatile("s_wait_asynccnt 0" ::: "memory");
    }
    __syncthreads();

    // S = Q * K^T for two 16-wide m sub-tiles (K operands from global, L1-hot).
    const su8* kp0 = (const su8*)(Kbb + (size_t)(m0 + lr) * CQ_);
    const su8* kp1 = (const su8*)(Kbb + (size_t)(m0 + 16 + lr) * CQ_);
    const v16bf kb0 = pack16(kp0[hi16 ? 2 : 0], kp0[hi16 ? 3 : 1]);
    const v16bf kb1 = pack16(kp1[hi16 ? 2 : 0], kp1[hi16 ? 3 : 1]);

    v8f z;
#pragma unroll
    for (int j = 0; j < 8; ++j) z[j] = 0.0f;
    v8f s0 = wmma_bf16(qa, kb0, z);
    v8f s1 = wmma_bf16(qa, kb1, z);

    // Online softmax per query row (row j / j+8 lives in lanes 0-15 / 16-31).
#pragma unroll
    for (int j = 0; j < 8; ++j) {
      float m2 = fmaxf(s0[j], s1[j]);
      m2 = fmaxf(m2, __shfl_xor(m2, 1, 32));
      m2 = fmaxf(m2, __shfl_xor(m2, 2, 32));
      m2 = fmaxf(m2, __shfl_xor(m2, 4, 32));
      m2 = fmaxf(m2, __shfl_xor(m2, 8, 32));
      const float mn = fmaxf(mx[j], m2);
      const float sc = __expf(mx[j] - mn);
      mx[j] = mn;
      const float p0 = __expf(s0[j] - mn);
      const float p1 = __expf(s1[j] - mn);
      lp[j] = lp[j] * sc + p0 + p1;
      const int row = j + (hi16 ? 8 : 0);
      Pbuf[w][row][lr]      = f2bfu(p0);
      Pbuf[w][row][16 + lr] = f2bfu(p1);
      if (lane == 0)  rowbuf[w][j]     = sc;   // broadcast scale by row index
      if (lane == 16) rowbuf[w][8 + j] = sc;
    }

    // Rescale O^T (column n = lane&15 -> per-lane row scale).
    const float csc = rowbuf[w][lr];
#pragma unroll
    for (int t = 0; t < 16; ++t)
#pragma unroll
      for (int j = 0; j < 8; ++j) O[t][j] *= csc;

    // P^T B-operand from LDS (row n, contiguous m).
    const su8* pp = (const su8*)&Pbuf[w][lr][0];
    const v16bf pB = pack16(pp[hi16 ? 2 : 0], pp[hi16 ? 3 : 1]);

    // O^T[c-tile] += V[16c x 32m] * P^T[32m x 16n]  -- 16 WMMAs per block.
#pragma unroll
    for (int t = 0; t < 16; ++t) {
      const su8* vp = (const su8*)&Vbuf[cur][t * 16 + lr][0];
      const v16bf vA = pack16(vp[hi16 ? 2 : 0], vp[hi16 ? 3 : 1]);
      O[t] = wmma_bf16(vA, pB, O[t]);
    }
    __syncthreads();   // everyone done reading Vbuf[cur] before it is refilled
  }

  // Final row-sum reduction, normalization, gamma * O + x (coalesced stores).
#pragma unroll
  for (int j = 0; j < 8; ++j) {
    float l = lp[j];
    l += __shfl_xor(l, 1, 32);
    l += __shfl_xor(l, 2, 32);
    l += __shfl_xor(l, 4, 32);
    l += __shfl_xor(l, 8, 32);
    const float li = 1.0f / l;
    if (lane == 0)  rowbuf[w][j]     = li;
    if (lane == 16) rowbuf[w][8 + j] = li;
  }
  const float linv = rowbuf[w][lr];
  const float g = gamma[0];
  const float* xb = x   + (size_t)b * C_ * N_;
  float*       ob = out + (size_t)b * C_ * N_;
#pragma unroll
  for (int t = 0; t < 16; ++t) {
#pragma unroll
    for (int j = 0; j < 8; ++j) {
      const int c = t * 16 + j + (hi16 ? 8 : 0);
      const size_t idx = (size_t)c * N_ + (n0 + lr);
      ob[idx] = g * O[t][j] * linv + xb[idx];
    }
  }
}

// ---------------------------------------------------------------------------
extern "C" void kernel_launch(void* const* d_in, const int* in_sizes, int n_in,
                              void* d_out, int out_size, void* d_ws, size_t ws_size,
                              hipStream_t stream) {
  const float* x     = (const float*)d_in[0];
  const float* Wq    = (const float*)d_in[1];
  const float* bq    = (const float*)d_in[2];
  const float* Wk    = (const float*)d_in[3];
  const float* bk    = (const float*)d_in[4];
  const float* Wv    = (const float*)d_in[5];
  const float* bv    = (const float*)d_in[6];
  const float* gamma = (const float*)d_in[7];
  float* out = (float*)d_out;

  // Workspace: Wc (160 KB) + Q (1 MB) + K (1 MB) + V (8 MB), all bf16. ~10.7 MB.
  unsigned short* Wcb = (unsigned short*)d_ws;
  unsigned short* Qb  = Wcb + (size_t)DTOT * C_;
  unsigned short* Kb  = Qb + (size_t)B_ * N_ * CQ_;
  unsigned short* Vb  = Kb + (size_t)B_ * N_ * CQ_;

  // Stage 0: weights -> bf16 (one-shot, tiny).
  wcvt_kernel<<<80, 256, 0, stream>>>(Wq, Wk, Wv, Wcb);
  // Stage 1: QKV projection. One WG per (batch, 64-wide n chunk).
  qkv_proj_kernel<<<256, 256, 0, stream>>>(x, bq, bk, bv, Wcb, Qb, Kb, Vb);
  // Stage 2: fused attention. 1024 query tiles / 8 waves per block.
  attn_kernel<<<128, 256, 0, stream>>>(x, gamma, Qb, Kb, Vb, out);
}
